// ModelName_86242943303934
// MI455X (gfx1250) — matrix operand order repaired and checked by reading, not compiled, not run on hardware
//
#include <hip/hip_runtime.h>
#include <math.h>
#include <stdint.h>

// ---------------------------------------------------------------------------
// CDNA5 (gfx1250) wave32 implementation of the hypergraph group-recsys
// forward pass.  Heavy GEMMs: v_wmma_f32_16x16x32_bf16 with a double-buffered
// LDS pipeline; A tiles DMA'd by the Tensor Data Mover (tensor_load_to_lds +
// s_wait_tensorcnt) overlapped with compute on the previous tile.
// ---------------------------------------------------------------------------

typedef __attribute__((ext_vector_type(16))) __bf16    v16bf;
typedef __attribute__((ext_vector_type(8)))  float     v8f;
typedef __attribute__((ext_vector_type(8)))  unsigned  v8u;
typedef __attribute__((ext_vector_type(4)))  unsigned  uint4v;
typedef __attribute__((ext_vector_type(8)))  int       int8v;
typedef __attribute__((ext_vector_type(4)))  int       int4v;

static constexpr int D = 128;   // embedding width (N dim of every GEMM)

#if __has_builtin(__builtin_amdgcn_tensor_load_to_lds) && \
    __has_builtin(__builtin_amdgcn_s_wait_tensorcnt)
#define HAVE_TDM 1
#else
#define HAVE_TDM 0
#endif

__device__ __forceinline__ __bf16 f2bf(float f) {
    unsigned u = __builtin_bit_cast(unsigned, f);
    unsigned r = u + 0x7FFFu + ((u >> 16) & 1u);           // round-nearest-even
    return __builtin_bit_cast(__bf16, (unsigned short)(r >> 16));
}
__device__ __forceinline__ float bf2f(__bf16 b) {
    unsigned u = ((unsigned)__builtin_bit_cast(unsigned short, b)) << 16;
    return __builtin_bit_cast(float, u);
}
__device__ __forceinline__ __bf16 bflo(unsigned v) {
    return __builtin_bit_cast(__bf16, (unsigned short)(v & 0xFFFFu));
}
__device__ __forceinline__ __bf16 bfhi(unsigned v) {
    return __builtin_bit_cast(__bf16, (unsigned short)(v >> 16));
}

// ---------------------------------------------------------------------------
// Tensor Data Mover: async 2D bf16 tile load (global -> LDS).
// g1w0 packs workgroup_mask/data_size/pad fields (D# group1 bits 31:0).
// ---------------------------------------------------------------------------
__device__ __forceinline__ void tdm_load_2d(unsigned lds_off, const void* gptr,
                                            unsigned tensor_d0, unsigned tensor_d1,
                                            unsigned tile_d0, unsigned tile_d1,
                                            unsigned d0_stride, unsigned g1w0)
{
#if HAVE_TDM
    unsigned long long ga = (unsigned long long)gptr;
    uint4v g0;
    g0[0] = 1u;                                           // count=1, user desc
    g0[1] = lds_off;                                      // lds_addr
    g0[2] = (unsigned)(ga & 0xFFFFFFFFull);               // global_addr lo
    g0[3] = (unsigned)((ga >> 32) & 0x01FFFFFFull)        // global_addr hi (57b)
          | (2u << 30);                                   // type = 2 (image)
    int8v g1;
    g1[0] = (int)g1w0;                                    // mask/ds/pad flags
    g1[1] = (int)((tensor_d0 & 0xFFFFu) << 16);           // tensor_dim0 lo16
    g1[2] = (int)((tensor_d0 >> 16) | ((tensor_d1 & 0xFFFFu) << 16));
    g1[3] = (int)((tensor_d1 >> 16) | (tile_d0 << 16));   // tile_dim0
    g1[4] = (int)(tile_d1 & 0xFFFFu);                     // tile_dim1, tile_dim2=0
    g1[5] = (int)d0_stride;                               // tensor_dim0_stride lo
    g1[6] = 0;
    g1[7] = 0;
    int4v z4 = {0, 0, 0, 0};
#if defined(__clang_major__) && (__clang_major__ >= 23)
    int8v z8 = {0, 0, 0, 0, 0, 0, 0, 0};
    __builtin_amdgcn_tensor_load_to_lds(g0, g1, z4, z4, z8, 0);
#else
    __builtin_amdgcn_tensor_load_to_lds(g0, g1, z4, z4, 0);
#endif
#else
    (void)lds_off; (void)gptr; (void)tensor_d0; (void)tensor_d1;
    (void)tile_d0; (void)tile_d1; (void)d0_stride; (void)g1w0;
#endif
}

// D# group1 word0 for bf16 tiles with LDS padding of 2 DWORDs every 16 DWORDs
// (row 32 halves=16 DWORDs -> padded stride 36 halves, matches lds_a layout):
// data_size=1(2B)<<16 | pad_enable<<20 | pad_interval=3(16dw)<<22 | pad_amount=1(2dw)<<25
#define TDM_G1W0_BF16_PAD36 0x02D10000u

// ---------------------------------------------------------------------------
// WMMA GEMM:  Y[m, 0..127] = (1/rowdeg[m]) * sum_k A[m,k] * X[k, 0..127]
//   TRANS=false: A = Hb            (Mout = Hrows, K = Hcols)
//   TRANS=true : A = Hb^T          (Mout = Hcols, K = Hrows)
// Hb is bf16 incidence (pre-converted); X is bf16 [K,128]; Y is bf16.
// Block: 256 threads = 8 waves; C tile 64 rows x 128 cols; K-step 32.
// Double-buffered LDS pipeline: tile i+1 staged (TDM for A in the
// non-transposed path) while tile i is consumed by the WMMAs.
// ---------------------------------------------------------------------------
template <bool TRANS>
__global__ __launch_bounds__(256) void hg_wmma_gemm(
    const __bf16* __restrict__ Hb, int Hrows, int Hcols,
    const __bf16* __restrict__ X,
    const float* __restrict__ rowdeg,
    __bf16* __restrict__ Y)
{
    const int Mout = TRANS ? Hcols : Hrows;
    const int K    = TRANS ? Hrows : Hcols;
    const int m0   = blockIdx.x * 64;

    // A tile [m][k] bf16, row stride 36 halves (TDM pad fields reproduce this).
    __shared__ __bf16 lds_a[2][64 * 36];
    // B tile transposed [n][k] bf16, row stride 36 halves.
    __shared__ __bf16 lds_bt[2][128 * 36];

    const int tid   = threadIdx.x;
    const int lane  = tid & 31;
    const int w     = tid >> 5;
    const int mw    = w & 3;
    const int nw    = w >> 2;
    const int mbase = mw * 16;
    const int nbase = nw * 64;
    const int lrow  = lane & 15;
    const int lh    = lane >> 4;          // lane half: 0 (0-15) / 1 (16-31)

    v8f acc[4] = {};

    const unsigned* __restrict__ Xu = (const unsigned*)X;   // bf16 pairs
    const unsigned* __restrict__ Hu = (const unsigned*)Hb;  // bf16 pairs

    // ---- tile staging helpers ------------------------------------------
    auto stage_a = [&](int buf, int k0) {
        if (!TRANS) {
#if HAVE_TDM
            // Async A-tile DMA: 64 rows x 32 K of Hb, row stride Hcols elems.
            if (w == 0) {
                tdm_load_2d((unsigned)(uintptr_t)(void*)&lds_a[buf][0],
                            Hb + (long)m0 * Hcols + k0,
                            (unsigned)(Hcols - k0), (unsigned)(Hrows - m0),
                            32u, 64u, (unsigned)Hcols,
                            TDM_G1W0_BF16_PAD36);
            }
#else
            // manual A: 64x32 halves = 1024 dwords
            if (m0 + 64 <= Hrows) {                      // fast path: no guards
                #pragma unroll
                for (int i = 0; i < 4; ++i) {
                    int idx = tid + i * 256;
                    int row = idx >> 4, c2 = idx & 15;
                    unsigned v = Hu[(long)(m0 + row) * (Hcols >> 1) + (k0 >> 1) + c2];
                    *(unsigned*)&lds_a[buf][row * 36 + 2 * c2] = v;
                }
            } else {
                #pragma unroll
                for (int i = 0; i < 4; ++i) {
                    int idx = tid + i * 256;
                    int row = idx >> 4, c2 = idx & 15;
                    int gm = m0 + row;
                    bool ok = (gm < Hrows);
                    long off = ok ? ((long)gm * (Hcols >> 1) + (k0 >> 1) + c2) : 0;
                    unsigned v = Hu[off];
                    v = ok ? v : 0u;
                    *(unsigned*)&lds_a[buf][row * 36 + 2 * c2] = v;
                }
            }
#endif
        } else {
            // manual A^T: read 32x64 halves as dwords, scatter-transpose.
            // Mout (=Hcols) is 64-aligned in all our uses; only K can be ragged.
            if (k0 + 32 <= Hrows) {                      // fast path: no guards
                #pragma unroll
                for (int i = 0; i < 4; ++i) {
                    int idx = tid + i * 256;
                    int kk = idx >> 5, c2 = idx & 31;
                    unsigned v = Hu[(long)(k0 + kk) * (Hcols >> 1) + (m0 >> 1) + c2];
                    lds_a[buf][(2 * c2)     * 36 + kk] = bflo(v);
                    lds_a[buf][(2 * c2 + 1) * 36 + kk] = bfhi(v);
                }
            } else {
                #pragma unroll
                for (int i = 0; i < 4; ++i) {
                    int idx = tid + i * 256;
                    int kk = idx >> 5, c2 = idx & 31;
                    int gk = k0 + kk;
                    bool ok = (gk < Hrows);
                    long off = ok ? ((long)gk * (Hcols >> 1) + (m0 >> 1) + c2) : 0;
                    unsigned v = Hu[off];
                    v = ok ? v : 0u;
                    lds_a[buf][(2 * c2)     * 36 + kk] = bflo(v);
                    lds_a[buf][(2 * c2 + 1) * 36 + kk] = bfhi(v);
                }
            }
        }
    };
    auto stage_b = [&](int buf, int k0) {
        // B tile 32x128 bf16, staged transposed into [n][k]
        if (k0 + 32 <= K) {                              // fast path: no guards
            #pragma unroll
            for (int i = 0; i < 8; ++i) {
                int idx = tid + i * 256;
                int kk = idx >> 6, c2 = idx & 63;
                unsigned v = Xu[(long)(k0 + kk) * 64 + c2];
                lds_bt[buf][(2 * c2)     * 36 + kk] = bflo(v);
                lds_bt[buf][(2 * c2 + 1) * 36 + kk] = bfhi(v);
            }
        } else {
            #pragma unroll
            for (int i = 0; i < 8; ++i) {
                int idx = tid + i * 256;
                int kk = idx >> 6, c2 = idx & 63;
                int gk = k0 + kk;
                bool ok = (gk < K);
                long off = ok ? ((long)gk * 64 + c2) : 0;
                unsigned v = Xu[off];
                v = ok ? v : 0u;
                lds_bt[buf][(2 * c2)     * 36 + kk] = bflo(v);
                lds_bt[buf][(2 * c2 + 1) * 36 + kk] = bfhi(v);
            }
        }
    };

    // ---- pipeline prologue: stage tile 0 -------------------------------
    stage_a(0, 0);
    stage_b(0, 0);
#if HAVE_TDM
    if (!TRANS) __builtin_amdgcn_s_wait_tensorcnt(0);
#endif
    __syncthreads();

    // ---- main pipelined loop -------------------------------------------
    for (int k0 = 0; k0 < K; k0 += 32) {
        const int cur = (k0 >> 5) & 1;
        const int nxt = cur ^ 1;
        const int k1  = k0 + 32;

        // stage tile i+1 while computing tile i (TDM overlaps with WMMA)
        if (k1 < K) {
            stage_a(nxt, k1);
            stage_b(nxt, k1);
            if (TRANS && lane == 0 && k1 + 32 < Hrows)
                __builtin_prefetch(&Hu[(long)(k1 + 32) * (Hcols >> 1) + (m0 >> 1)], 0, 1);
        }

        // ---- A fragment (16x32 bf16, wave32 layout) --------------------
        // lanes 0-15: K={0..7,16..23}; lanes 16-31: K={8..15,24..31}
        const unsigned* apu = (const unsigned*)&lds_a[cur][(mbase + lrow) * 36];
        const int ka2 = lh ? 4 : 0;
        v8u au;
        #pragma unroll
        for (int j = 0; j < 4; ++j) au[j]     = apu[ka2 + j];
        #pragma unroll
        for (int j = 0; j < 4; ++j) au[4 + j] = apu[8 + ka2 + j];
        v16bf a = __builtin_bit_cast(v16bf, au);

        // ---- B fragments + WMMA ----------------------------------------
        const int kb2 = lh ? 4 : 0;
        #pragma unroll
        for (int nt = 0; nt < 4; ++nt) {
            const unsigned* bpu =
                (const unsigned*)&lds_bt[cur][(nbase + nt * 16 + lrow) * 36];
            v8u bu;
            #pragma unroll
            for (int j = 0; j < 8; ++j) bu[j] = bpu[kb2 + (j < 4 ? j : j + 4)];
            v16bf b = __builtin_bit_cast(v16bf, bu);
            acc[nt] = __builtin_amdgcn_wmma_f32_16x16x32_bf16(
                false, a, false, b, (short)0, acc[nt], false, false);
        }

#if HAVE_TDM
        if (!TRANS) __builtin_amdgcn_s_wait_tensorcnt(0);   // next A tile landed
#endif
        __syncthreads();   // publish tile i+1; reads of tile i are complete
    }

    // ---- epilogue: per-row 1/deg scale, write bf16 ----------------------
    #pragma unroll
    for (int nt = 0; nt < 4; ++nt) {
        #pragma unroll
        for (int r = 0; r < 8; ++r) {
            int row = m0 + mbase + lh * 8 + r;       // C layout: VGPR r -> M=r / 8+r
            int col = nbase + nt * 16 + lrow;
            if (row < Mout) {
                float v = acc[nt][r] / rowdeg[row];
                Y[(long)row * D + col] = f2bf(v);
            }
        }
    }
}

// ---------------------------------------------------------------------------
// Fused prep: stream H (fp32) once (non-temporal) -> bf16 copy + column sums.
// Thread-per-column over a row chunk; fully coalesced read and write.
// ---------------------------------------------------------------------------
__global__ void prep_H_kernel(const float* __restrict__ H, int Mrows, int N,
                              int rows_per_block,
                              __bf16* __restrict__ Hb, float* __restrict__ colsum)
{
    int c = blockIdx.x * blockDim.x + threadIdx.x;
    if (c >= N) return;
    int r0 = blockIdx.y * rows_per_block;
    int r1 = r0 + rows_per_block; if (r1 > Mrows) r1 = Mrows;
    float s = 0.f;
    for (int r = r0; r < r1; ++r) {
        float v = __builtin_nontemporal_load(&H[(long)r * N + c]);
        Hb[(long)r * N + c] = f2bf(v);
        s += v;
    }
    atomicAdd(&colsum[c], s);
}

// Row sums over the bf16 copy (exact: 0/1 values): wave per row.
__global__ void row_sums_bf16_kernel(const __bf16* __restrict__ Hb, int M, int N,
                                     float* __restrict__ out)
{
    int row = blockIdx.x * (blockDim.x >> 5) + (threadIdx.x >> 5);
    if (row >= M) return;
    int lane = threadIdx.x & 31;
    const unsigned* p = (const unsigned*)(Hb + (long)row * N);
    float s = 0.f;
    for (int c2 = lane; c2 < (N >> 1); c2 += 32) {
        unsigned v = p[c2];
        s += bf2f(bflo(v)) + bf2f(bfhi(v));
    }
    #pragma unroll
    for (int off = 16; off; off >>= 1) s += __shfl_down(s, off, 32);
    if (lane == 0) out[row] = s + 1e-5f;
}

__global__ void fill_kernel(float* __restrict__ p, float val, int n)
{
    int i = blockIdx.x * blockDim.x + threadIdx.x;
    if (i < n) p[i] = val;
}

// ---------------------------------------------------------------------------
// dtype glue
// ---------------------------------------------------------------------------
__global__ void f32_to_bf16_kernel(const float* __restrict__ x,
                                   __bf16* __restrict__ y, int n)
{
    int i = blockIdx.x * blockDim.x + threadIdx.x;
    int stride = gridDim.x * blockDim.x;
    for (; i < n; i += stride) y[i] = f2bf(x[i]);
}

__global__ void avg_bf16_to_f32_kernel(const __bf16* __restrict__ a,
                                       const __bf16* __restrict__ b,
                                       float* __restrict__ y, int n)
{
    int i = blockIdx.x * blockDim.x + threadIdx.x;
    int stride = gridDim.x * blockDim.x;
    for (; i < n; i += stride) y[i] = 0.5f * (bf2f(a[i]) + bf2f(b[i]));
}

// ---------------------------------------------------------------------------
// Attention: one wave per member.  Lanes 0..15 each compute one hidden unit
// of Linear(256,16)+ReLU, then shuffle-reduce the Linear(16,1) logit.
// ---------------------------------------------------------------------------
__global__ void att_logit_kernel(const float* __restrict__ user,
                                 const float* __restrict__ item_emb,
                                 const int* __restrict__ member_uid,
                                 const int* __restrict__ batch_seg,
                                 const int* __restrict__ itemid,
                                 const float* __restrict__ w1,
                                 const float* __restrict__ b1,
                                 const float* __restrict__ w2,
                                 const float* __restrict__ b2,
                                 float* __restrict__ att,
                                 float* __restrict__ denom, int M)
{
    int i = blockIdx.x * (blockDim.x >> 5) + (threadIdx.x >> 5);
    if (i >= M) return;
    int lane = threadIdx.x & 31;
    int uid = member_uid[i];
    int seg = batch_seg[i];
    int iid = itemid[seg];
    const float* u  = user + (long)uid * D;
    const float* it = item_emb + (long)iid * D;
    float h = 0.f;
    if (lane < 16) {
        for (int d = 0; d < D; ++d) h = fmaf(u[d],  w1[d * 16 + lane], h);
        for (int d = 0; d < D; ++d) h = fmaf(it[d], w1[(D + d) * 16 + lane], h);
        h += b1[lane];
        h = fmaxf(h, 0.f) * w2[lane];
    }
    #pragma unroll
    for (int off = 16; off; off >>= 1) h += __shfl_down(h, off, 32);
    if (lane == 0) {
        float e = __expf(h + b2[0]);
        att[i] = e;
        atomicAdd(&denom[seg], e);
    }
}

__global__ void att_pool_kernel(const float* __restrict__ user,
                                const int* __restrict__ member_uid,
                                const int* __restrict__ batch_seg,
                                const float* __restrict__ att,
                                const float* __restrict__ denom,
                                float* __restrict__ affect, int M)
{
    int i = blockIdx.x * (blockDim.x >> 5) + (threadIdx.x >> 5);
    if (i >= M) return;
    int lane = threadIdx.x & 31;
    int uid = member_uid[i];
    int seg = batch_seg[i];
    float wgt = att[i] / denom[seg];
    const float* u = user + (long)uid * D;
    float* dst = affect + (long)seg * D;
    #pragma unroll
    for (int d = lane; d < D; d += 32) atomicAdd(&dst[d], wgt * u[d]);
}

// ---------------------------------------------------------------------------
// Final prediction MLP + sigmoid: one wave per sample; lanes 0..7 compute
// hidden units of Linear(384,8)+ReLU, shuffle-reduce Linear(8,1).
// ---------------------------------------------------------------------------
__global__ void predict_kernel(const float* __restrict__ affect,
                               const __bf16* __restrict__ g2,
                               const float* __restrict__ item_emb,
                               const int* __restrict__ groupid,
                               const int* __restrict__ itemid,
                               const float* __restrict__ w1,
                               const float* __restrict__ b1,
                               const float* __restrict__ w2,
                               const float* __restrict__ b2,
                               float* __restrict__ out, int B)
{
    int bi = blockIdx.x * (blockDim.x >> 5) + (threadIdx.x >> 5);
    if (bi >= B) return;
    int lane = threadIdx.x & 31;
    int gid = groupid[bi];
    int iid = itemid[bi];
    const float*  af = affect   + (long)bi  * D;
    const __bf16* ge = g2       + (long)gid * D;
    const float*  it = item_emb + (long)iid * D;
    float h = 0.f;
    if (lane < 8) {
        for (int d = 0; d < D; ++d) {
            float g  = af[d] + bf2f(ge[d]);
            float iv = it[d];
            h = fmaf(g * iv, w1[d * 8 + lane],           h);   // group*item
            h = fmaf(g,      w1[(D + d) * 8 + lane],     h);   // group
            h = fmaf(iv,     w1[(2 * D + d) * 8 + lane], h);   // item
        }
        h += b1[lane];
        h = fmaxf(h, 0.f) * w2[lane];
    }
    #pragma unroll
    for (int off = 16; off; off >>= 1) h += __shfl_down(h, off, 32);
    if (lane == 0) out[bi] = 1.f / (1.f + __expf(-(h + b2[0])));
}

// ---------------------------------------------------------------------------
// Orchestration
// ---------------------------------------------------------------------------
extern "C" void kernel_launch(void* const* d_in, const int* in_sizes, int n_in,
                              void* d_out, int out_size, void* d_ws, size_t ws_size,
                              hipStream_t stream)
{
    const float* user_emb  = (const float*)d_in[0];
    const float* item_emb  = (const float*)d_in[1];
    const float* group_emb = (const float*)d_in[2];
    const float* H_ug      = (const float*)d_in[3];
    const float* H_af      = (const float*)d_in[4];
    const float* H_gg      = (const float*)d_in[5];
    const float* att_w1    = (const float*)d_in[6];
    const float* att_b1    = (const float*)d_in[7];
    const float* att_w2    = (const float*)d_in[8];
    const float* att_b2    = (const float*)d_in[9];
    const float* pred_w1   = (const float*)d_in[10];
    const float* pred_b1   = (const float*)d_in[11];
    const float* pred_w2   = (const float*)d_in[12];
    const float* pred_b2   = (const float*)d_in[13];
    const int*   groupid   = (const int*)d_in[14];
    const int*   itemid    = (const int*)d_in[15];
    const int*   member_uid= (const int*)d_in[16];
    const int*   batch_seg = (const int*)d_in[17];

    const int U = in_sizes[0] / D;    // 30000
    const int G = in_sizes[2] / D;    // 4096
    const int B = in_sizes[14];       // 2048
    const int M = in_sizes[16];       // total members in batch

    // ---- workspace carve-out -------------------------------------------
    char* wsp = (char*)d_ws;
    auto alloc = [&](size_t bytes) -> char* {
        char* p = wsp;
        wsp += (bytes + 255) & ~(size_t)255;
        return p;
    };
    __bf16* Hb    = (__bf16*)alloc((size_t)U * G * 2);  // bf16 H (reused x3)
    __bf16* x0    = (__bf16*)alloc((size_t)U * D * 2);  // user_emb in bf16
    __bf16* u1    = (__bf16*)alloc((size_t)U * D * 2);  // hg_prop(H_ug)
    __bf16* u2    = (__bf16*)alloc((size_t)U * D * 2);  // hg_prop(H_ug_affect)
    __bf16* tbuf  = (__bf16*)alloc((size_t)G * D * 2);  // edge-space temp
    __bf16* g0    = (__bf16*)alloc((size_t)G * D * 2);  // group_emb in bf16
    __bf16* g1    = (__bf16*)alloc((size_t)G * D * 2);
    __bf16* g2b   = (__bf16*)alloc((size_t)G * D * 2);  // hg_prop(H_gg) result
    float*  userf = (float*)alloc((size_t)U * D * 4);   // 0.5*(u1+u2)
    float*  dv    = (float*)alloc((size_t)U * 4);       // vertex degrees (reused)
    float*  de    = (float*)alloc((size_t)G * 4);       // edge degrees (reused)
    float*  attv  = (float*)alloc((size_t)M * 4);
    float*  denom = (float*)alloc((size_t)B * 4);
    float*  affect= (float*)alloc((size_t)B * D * 4);
    (void)ws_size; (void)n_in; (void)out_size;

    // ---- embeddings -> bf16 --------------------------------------------
    f32_to_bf16_kernel<<<1024, 256, 0, stream>>>(user_emb,  x0, U * D);
    f32_to_bf16_kernel<<<256,  256, 0, stream>>>(group_emb, g0, G * D);

    const int RPB    = 256;
    const int gridMu = (U + 63) / 64;
    const int gridMg = (G + 63) / 64;
    const dim3 prepU((G + 255) / 256, (U + RPB - 1) / RPB);
    const dim3 prepG((G + 255) / 256, (G + RPB - 1) / RPB);

    // ==== H_ug: prep (fp32 -> bf16 + degrees), then hg_prop(·, x0, 2) ====
    fill_kernel<<<(G + 255) / 256, 256, 0, stream>>>(de, 1e-5f, G);
    prep_H_kernel<<<prepU, 256, 0, stream>>>(H_ug, U, G, RPB, Hb, de);
    row_sums_bf16_kernel<<<(U + 7) / 8, 256, 0, stream>>>(Hb, U, G, dv);
    hg_wmma_gemm<true ><<<gridMg, 256, 0, stream>>>(Hb, U, G, x0,   de, tbuf);
    hg_wmma_gemm<false><<<gridMu, 256, 0, stream>>>(Hb, U, G, tbuf, dv, u1);
    hg_wmma_gemm<true ><<<gridMg, 256, 0, stream>>>(Hb, U, G, u1,   de, tbuf);
    hg_wmma_gemm<false><<<gridMu, 256, 0, stream>>>(Hb, U, G, tbuf, dv, u1);

    // ==== H_ug_affect ====================================================
    fill_kernel<<<(G + 255) / 256, 256, 0, stream>>>(de, 1e-5f, G);
    prep_H_kernel<<<prepU, 256, 0, stream>>>(H_af, U, G, RPB, Hb, de);
    row_sums_bf16_kernel<<<(U + 7) / 8, 256, 0, stream>>>(Hb, U, G, dv);
    hg_wmma_gemm<true ><<<gridMg, 256, 0, stream>>>(Hb, U, G, x0,   de, tbuf);
    hg_wmma_gemm<false><<<gridMu, 256, 0, stream>>>(Hb, U, G, tbuf, dv, u2);
    hg_wmma_gemm<true ><<<gridMg, 256, 0, stream>>>(Hb, U, G, u2,   de, tbuf);
    hg_wmma_gemm<false><<<gridMu, 256, 0, stream>>>(Hb, U, G, tbuf, dv, u2);

    // ==== H_gg ===========================================================
    fill_kernel<<<(G + 255) / 256, 256, 0, stream>>>(de, 1e-5f, G);
    prep_H_kernel<<<prepG, 256, 0, stream>>>(H_gg, G, G, RPB, Hb, de);
    row_sums_bf16_kernel<<<(G + 7) / 8, 256, 0, stream>>>(Hb, G, G, dv);
    hg_wmma_gemm<true ><<<gridMg, 256, 0, stream>>>(Hb, G, G, g0,   de, tbuf);
    hg_wmma_gemm<false><<<gridMg, 256, 0, stream>>>(Hb, G, G, tbuf, dv, g1);
    hg_wmma_gemm<true ><<<gridMg, 256, 0, stream>>>(Hb, G, G, g1,   de, tbuf);
    hg_wmma_gemm<false><<<gridMg, 256, 0, stream>>>(Hb, G, G, tbuf, dv, g2b);

    // ---- user = 0.5*(u1+u2) as f32 -------------------------------------
    avg_bf16_to_f32_kernel<<<1024, 256, 0, stream>>>(u1, u2, userf, U * D);

    // ---- attention + segment softmax-pool ------------------------------
    fill_kernel<<<(B + 255) / 256, 256, 0, stream>>>(denom, 0.f, B);
    fill_kernel<<<(B * D + 255) / 256, 256, 0, stream>>>(affect, 0.f, B * D);
    att_logit_kernel<<<(M + 7) / 8, 256, 0, stream>>>(
        userf, item_emb, member_uid, batch_seg, itemid,
        att_w1, att_b1, att_w2, att_b2, attv, denom, M);
    att_pool_kernel<<<(M + 7) / 8, 256, 0, stream>>>(
        userf, member_uid, batch_seg, attv, denom, affect, M);

    // ---- prediction head ------------------------------------------------
    predict_kernel<<<(B + 7) / 8, 256, 0, stream>>>(
        affect, g2b, item_emb, groupid, itemid,
        pred_w1, pred_b1, pred_w2, pred_b2, (float*)d_out, B);
}
